// DifferentialMultiheadAttention_80573586473581
// MI455X (gfx1250) — matrix operand order repaired and checked by reading, not compiled, not run on hardware
//
#include <hip/hip_runtime.h>
#include <hip/hip_bf16.h>

// ---------------------------------------------------------------------------
// Differential Multi-head Attention for gfx1250 (MI455X), bf16 WMMA path.
//   EMBED_DIM=1024, NUM_HEADS=16, HEAD_DIM=64, B=2, N=2048, LAMBDA_INIT=0.8
// ---------------------------------------------------------------------------

#define EMBED  1024
#define SEQ    2048
#define BATCH  2
#define HEADS  16
#define HDIM   64
#define MROWS  (BATCH * SEQ)          // 4096 rows for all GEMMs

typedef __attribute__((ext_vector_type(16))) __bf16 v16bf;
typedef __attribute__((ext_vector_type(8)))  float  v8f;

union FragB16 { v16bf v; unsigned short u[16]; unsigned int u32[8]; };
union FragF   { v8f   v; float f[8]; };

__device__ __forceinline__ unsigned short f2bf(float x) {
    unsigned int u = __builtin_bit_cast(unsigned int, x);
    unsigned int r = u + 0x7FFFu + ((u >> 16) & 1u);   // round-to-nearest-even
    return (unsigned short)(r >> 16);
}

// A-matrix (16x32 bf16) per-lane K index, ISA 7.12.2: lane m=L&15, half=L>>4
__device__ __forceinline__ int ka_idx(int v, int half) {
    return ((v & 3) * 2) + half * 8 + ((v >> 2) * 16);  // +j for the pair
}
// B-matrix (32x16 bf16) per-lane K index: lane n=L&15, half=L>>4
__device__ __forceinline__ int kb_idx(int v, int half) {
    return ((v & 3) * 2) + half * 8 + ((v >> 2) * 16);  // +j for the pair
}

// xor-butterfly within 16-lane halves via ds_swizzle (group-of-32 mode:
// new_lane = ((lane & 0x1F) | 0) ^ xor_mask ; masks < 16 never cross halves)
template <int IMM>
__device__ __forceinline__ float swz(float x) {
    int i = __builtin_bit_cast(int, x);
    i = __builtin_amdgcn_ds_swizzle(i, IMM);
    return __builtin_bit_cast(float, i);
}
__device__ __forceinline__ float redmax16(float x) {
    x = fmaxf(x, swz<0x041F>(x));   // xor 1
    x = fmaxf(x, swz<0x081F>(x));   // xor 2
    x = fmaxf(x, swz<0x101F>(x));   // xor 4
    x = fmaxf(x, swz<0x201F>(x));   // xor 8
    return x;
}
__device__ __forceinline__ float redsum16(float x) {
    x += swz<0x041F>(x);
    x += swz<0x081F>(x);
    x += swz<0x101F>(x);
    x += swz<0x201F>(x);
    return x;
}

// ---------------------------------------------------------------------------
// lambda scalar: lv = mean_h( exp(min(<lq1,lk1>,5)) - exp(min(<lq2,lk2>,5)) + 0.8 )
// ---------------------------------------------------------------------------
__global__ void lambda_kernel(const float* lq1, const float* lk1,
                              const float* lq2, const float* lk2, float* lv) {
    int lane = threadIdx.x;   // 32 threads
    float acc = 0.f;
    for (int h = 0; h < 8; ++h) {
        float s1 = 0.f, s2 = 0.f;
        for (int d = lane; d < HDIM; d += 32) {
            s1 += lq1[h * HDIM + d] * lk1[h * HDIM + d];
            s2 += lq2[h * HDIM + d] * lk2[h * HDIM + d];
        }
        for (int m = 16; m >= 1; m >>= 1) {
            s1 += __shfl_xor(s1, m, 32);
            s2 += __shfl_xor(s2, m, 32);
        }
        acc += __expf(fminf(s1, 5.f)) - __expf(fminf(s2, 5.f)) + 0.8f;
    }
    if (lane == 0) *lv = acc * 0.125f;
}

// ---------------------------------------------------------------------------
// f32 -> bf16 conversion (n multiple of 4)
// ---------------------------------------------------------------------------
__global__ void cvt_kernel(const float* __restrict__ in,
                           unsigned short* __restrict__ out, int n) {
    int i = (blockIdx.x * blockDim.x + threadIdx.x) * 4;
    if (i >= n) return;
    unsigned short r0 = f2bf(in[i + 0]);
    unsigned short r1 = f2bf(in[i + 1]);
    unsigned short r2 = f2bf(in[i + 2]);
    unsigned short r3 = f2bf(in[i + 3]);
    unsigned int lo = (unsigned int)r0 | ((unsigned int)r1 << 16);
    unsigned int hi = (unsigned int)r2 | ((unsigned int)r3 << 16);
    *(uint2*)&out[i] = make_uint2(lo, hi);
}

// ---------------------------------------------------------------------------
// GEMM: C[M,N] = A[M,K](bf16) * W[N,K]^T(bf16) + bias[N]
// Wave tile = 16(M) x 64(N): 1 A-frag reused across 4 B-frags per K-step.
// ---------------------------------------------------------------------------
template <int OUT_BF16>
__global__ void __launch_bounds__(256)
gemm16(const unsigned short* __restrict__ A, const unsigned short* __restrict__ W,
       const float* __restrict__ bias, void* __restrict__ Cout,
       int M, int N, int K) {
    const int lane = threadIdx.x & 31;
    const int wave = threadIdx.x >> 5;
    const int wid  = blockIdx.x * 8 + wave;
    const int tilesN = N >> 6;
    const int tm = wid / tilesN;
    const int tn = wid % tilesN;
    const int m    = lane & 15;
    const int half = lane >> 4;
    const int rowA = tm * 16 + m;

    FragF acc[4];
#pragma unroll
    for (int nt = 0; nt < 4; ++nt)
#pragma unroll
        for (int r = 0; r < 8; ++r) acc[nt].f[r] = 0.f;

    for (int kk = 0; kk < K; kk += 32) {
        FragB16 a;
#pragma unroll
        for (int v = 0; v < 8; ++v)
            a.u32[v] = *(const unsigned int*)&A[(size_t)rowA * K + kk + ka_idx(v, half)];
        __builtin_prefetch(&A[(size_t)rowA * K + kk + 64], 0, 1);  // global_prefetch_b8
#pragma unroll
        for (int nt = 0; nt < 4; ++nt) {
            FragB16 b;
            const int colW = tn * 64 + nt * 16 + m;
#pragma unroll
            for (int v = 0; v < 8; ++v)
                b.u32[v] = *(const unsigned int*)&W[(size_t)colW * K + kk + kb_idx(v, half)];
            acc[nt].v = __builtin_amdgcn_wmma_f32_16x16x32_bf16(
                false, a.v, false, b.v, (short)0, acc[nt].v, false, false);
        }
    }

#pragma unroll
    for (int nt = 0; nt < 4; ++nt)
#pragma unroll
        for (int r = 0; r < 8; ++r) {
            const int row = tm * 16 + r + 8 * half;
            const int col = tn * 64 + nt * 16 + m;
            const float val = acc[nt].f[r] + bias[col];
            if (OUT_BF16)
                ((unsigned short*)Cout)[(size_t)row * N + col] = f2bf(val);
            else
                ((float*)Cout)[(size_t)row * N + col] = val;
        }
}

// ---------------------------------------------------------------------------
// Flash attention. One block = 8 waves, all sharing (batch, head); each wave
// owns 16 queries. K/V tiles (32 keys) are staged cooperatively in LDS:
//   lds_k[key][d]        row-major      (b128 global load -> b128 ds store)
//   lds_v[d][key] (pad 8) transposed    (so AV B-frags are contiguous b128)
// Q/K/V stored as [B, N, 1024] bf16 with channel = h*64 + d.
// ---------------------------------------------------------------------------
#define VPAD 40   // 32 keys + 8 pad -> 80B row stride keeps 16B alignment

__global__ void __launch_bounds__(256)
attn_kernel(const unsigned short* __restrict__ Qb,
            const unsigned short* __restrict__ Kb,
            const unsigned short* __restrict__ Vb,
            float* __restrict__ Oraw) {
    __shared__ unsigned short lds_k[32][HDIM];       // 4 KB
    __shared__ unsigned short lds_v[HDIM][VPAD];     // 5 KB
    __shared__ unsigned short plds[8][16][32];       // 8 KB, per-wave P staging

    const int tid  = threadIdx.x;
    const int lane = tid & 31;
    const int w    = tid >> 5;
    const int wid  = blockIdx.x * 8 + w;
    const int qt   = wid & 127;          // query tile (2048/16 = 128)
    const int h    = (wid >> 7) & 15;    // head   (same for whole block)
    const int b    = wid >> 11;          // batch  (same for whole block)
    const int m    = lane & 15;
    const int half = lane >> 4;
    const float scale = 0.125f;          // 64^-0.5
    const size_t headOff = (size_t)h * HDIM;

    // cooperative staging assignment: 256 threads cover 32 keys x 8 d-chunks
    const int skey = tid >> 3;           // 0..31
    const int sd8  = (tid & 7) * 8;      // 0,8,...,56

    // Load Q as two 16x32 A-fragments (d = 0..31, 32..63)
    FragB16 qa[2];
    const int qrow = qt * 16 + m;
    const size_t qbase = ((size_t)(b * SEQ + qrow)) * EMBED + headOff;
#pragma unroll
    for (int s = 0; s < 2; ++s)
#pragma unroll
        for (int v = 0; v < 8; ++v)
            qa[s].u32[v] = *(const unsigned int*)&Qb[qbase + s * 32 + ka_idx(v, half)];

    FragF acc[4];
#pragma unroll
    for (int nt = 0; nt < 4; ++nt)
#pragma unroll
        for (int r = 0; r < 8; ++r) acc[nt].f[r] = 0.f;
    float mrow[8], lrow[8];
#pragma unroll
    for (int r = 0; r < 8; ++r) { mrow[r] = -1e30f; lrow[r] = 0.f; }

    for (int j0 = 0; j0 < SEQ; j0 += 32) {
        // ---- cooperative K/V tile staging (b128 loads) ----
        {
            const size_t rowoff = ((size_t)(b * SEQ + j0 + skey)) * EMBED + headOff + sd8;
            uint4 kq = *(const uint4*)&Kb[rowoff];
            *(uint4*)&lds_k[skey][sd8] = kq;
            union { uint4 q; unsigned short us[8]; } uv;
            uv.q = *(const uint4*)&Vb[rowoff];
#pragma unroll
            for (int i = 0; i < 8; ++i)
                lds_v[sd8 + i][skey] = uv.us[i];      // transpose into LDS
        }
        __syncthreads();

        // ---- S = Q @ K^T over this 32-key tile (two 16-key score frags) ----
        FragF S[2];
#pragma unroll
        for (int t = 0; t < 2; ++t) {
#pragma unroll
            for (int r = 0; r < 8; ++r) S[t].f[r] = 0.f;
            const int keyr = t * 16 + m;              // lane's B column (local key)
#pragma unroll
            for (int s = 0; s < 2; ++s) {
                FragB16 kb;
#pragma unroll
                for (int v = 0; v < 8; ++v)
                    kb.u32[v] = *(const unsigned int*)&lds_k[keyr][s * 32 + kb_idx(v, half)];
                S[t].v = __builtin_amdgcn_wmma_f32_16x16x32_bf16(
                    false, qa[s].v, false, kb.v, (short)0, S[t].v, false, false);
            }
        }

        // ---- online softmax (rows r+8*half live across lanes of one half) ----
#pragma unroll
        for (int r = 0; r < 8; ++r) {
            float s0 = S[0].f[r] * scale;
            float s1 = S[1].f[r] * scale;
            float mx = redmax16(fmaxf(s0, s1));
            const float mnew  = fmaxf(mrow[r], mx);
            const float alpha = __expf(mrow[r] - mnew);
            const float p0 = __expf(s0 - mnew);
            const float p1 = __expf(s1 - mnew);
            const float rs = redsum16(p0 + p1);
            lrow[r] = lrow[r] * alpha + rs;
            mrow[r] = mnew;
#pragma unroll
            for (int nt = 0; nt < 4; ++nt) acc[nt].f[r] *= alpha;
            // stage P into LDS (C-frag layout -> row-major 16x32 bf16 tile)
            plds[w][r + 8 * half][m]      = f2bf(p0);
            plds[w][r + 8 * half][16 + m] = f2bf(p1);
        }
        asm volatile("s_wait_dscnt 0" ::: "memory");

        // reload P as an A-fragment (16x32)
        FragB16 pa;
#pragma unroll
        for (int v = 0; v < 8; ++v)
            pa.u32[v] = *(const unsigned int*)&plds[w][m][ka_idx(v, half)];

        // ---- acc += P @ V (4 d-tiles of 16); B-frags from transposed LDS ----
#pragma unroll
        for (int nt = 0; nt < 4; ++nt) {
            FragB16 vb;
            const int d = nt * 16 + m;                // lane's B column
#pragma unroll
            for (int v = 0; v < 8; ++v)               // key = ka_idx(v,half)+j
                vb.u32[v] = *(const unsigned int*)&lds_v[d][ka_idx(v, half)];
            acc[nt].v = __builtin_amdgcn_wmma_f32_16x16x32_bf16(
                false, pa.v, false, vb.v, (short)0, acc[nt].v, false, false);
        }
        __syncthreads();   // protect lds_k / lds_v before next staging
    }

    // ---- epilogue: out = acc / l ----
#pragma unroll
    for (int nt = 0; nt < 4; ++nt)
#pragma unroll
        for (int r = 0; r < 8; ++r) {
            const int q = qt * 16 + r + 8 * half;
            const int d = nt * 16 + m;
            Oraw[((size_t)(b * SEQ + q)) * EMBED + headOff + d] =
                acc[nt].f[r] / lrow[r];
        }
}

// ---------------------------------------------------------------------------
// combine: y[:, :, c<512] = x1 - lv * x2 ; y[:, :, c>=512] = x2   (bf16 out)
// araw channel = h*64 + d, heads 0..7 -> x1, heads 8..15 -> x2
// ---------------------------------------------------------------------------
__global__ void combine_kernel(const float* __restrict__ araw,
                               const float* __restrict__ lvp,
                               unsigned short* __restrict__ yb, int total) {
    int i = blockIdx.x * blockDim.x + threadIdx.x;
    if (i >= total) return;
    const int c = i & (EMBED - 1);
    const float lv = *lvp;
    float val;
    if (c < EMBED / 2) val = araw[i] - lv * araw[i + EMBED / 2];
    else               val = araw[i];
    yb[i] = f2bf(val);
}

// ---------------------------------------------------------------------------
// launch
// ---------------------------------------------------------------------------
extern "C" void kernel_launch(void* const* d_in, const int* in_sizes, int n_in,
                              void* d_out, int out_size, void* d_ws, size_t ws_size,
                              hipStream_t stream) {
    (void)in_sizes; (void)n_in; (void)out_size; (void)ws_size;

    const float* x   = (const float*)d_in[0];
    const float* Wq  = (const float*)d_in[1];
    const float* bq  = (const float*)d_in[2];
    const float* Wk  = (const float*)d_in[3];
    const float* bk  = (const float*)d_in[4];
    const float* Wv  = (const float*)d_in[5];
    const float* bv  = (const float*)d_in[6];
    const float* Wp  = (const float*)d_in[7];
    const float* bp  = (const float*)d_in[8];
    const float* lq1 = (const float*)d_in[9];
    const float* lk1 = (const float*)d_in[10];
    const float* lq2 = (const float*)d_in[11];
    const float* lk2 = (const float*)d_in[12];

    char* ws = (char*)d_ws;
    size_t off = 0;
    float* lv = (float*)(ws + off);              off += 256;
    const size_t XB   = (size_t)MROWS * EMBED * sizeof(unsigned short);  // 8 MB
    const size_t WB   = (size_t)EMBED * EMBED * sizeof(unsigned short);  // 2 MB
    unsigned short* xb  = (unsigned short*)(ws + off); off += XB;
    unsigned short* wqb = (unsigned short*)(ws + off); off += WB;
    unsigned short* wkb = (unsigned short*)(ws + off); off += WB;
    unsigned short* wvb = (unsigned short*)(ws + off); off += WB;
    unsigned short* wpb = (unsigned short*)(ws + off); off += WB;
    unsigned short* qb  = (unsigned short*)(ws + off); off += XB;
    unsigned short* kb  = (unsigned short*)(ws + off); off += XB;
    unsigned short* vb  = (unsigned short*)(ws + off); off += XB;
    float*          araw = (float*)(ws + off);         off += (size_t)MROWS * EMBED * sizeof(float);
    unsigned short* yb  = (unsigned short*)(ws + off); off += XB;

    // 1. lambda scalar
    lambda_kernel<<<1, 32, 0, stream>>>(lq1, lk1, lq2, lk2, lv);

    // 2. bf16 conversions
    const int NX = MROWS * EMBED;   // 4194304
    const int NW = EMBED * EMBED;   // 1048576
    cvt_kernel<<<NX / 4 / 256, 256, 0, stream>>>(x,  xb,  NX);
    cvt_kernel<<<NW / 4 / 256, 256, 0, stream>>>(Wq, wqb, NW);
    cvt_kernel<<<NW / 4 / 256, 256, 0, stream>>>(Wk, wkb, NW);
    cvt_kernel<<<NW / 4 / 256, 256, 0, stream>>>(Wv, wvb, NW);
    cvt_kernel<<<NW / 4 / 256, 256, 0, stream>>>(Wp, wpb, NW);

    // 3. QKV projections (bf16 out)
    const int gemmBlocks = (MROWS / 16) * (EMBED / 64) / 8;  // 512
    gemm16<1><<<gemmBlocks, 256, 0, stream>>>(xb, wqb, bq, qb, MROWS, EMBED, EMBED);
    gemm16<1><<<gemmBlocks, 256, 0, stream>>>(xb, wkb, bk, kb, MROWS, EMBED, EMBED);
    gemm16<1><<<gemmBlocks, 256, 0, stream>>>(xb, wvb, bv, vb, MROWS, EMBED, EMBED);

    // 4. attention: 2 batch * 16 heads * 128 q-tiles = 4096 waves / 8 = 512 blocks
    attn_kernel<<<512, 256, 0, stream>>>(qb, kb, vb, araw);

    // 5. differential combine -> bf16
    combine_kernel<<<NX / 256, 256, 0, stream>>>(araw, lv, yb, NX);

    // 6. output projection (f32 out)
    gemm16<0><<<gemmBlocks, 256, 0, stream>>>(yb, wpb, bp, d_out, MROWS, EMBED, EMBED);
}
